// SparseMoE_9929964388838
// MI455X (gfx1250) — matrix operand order repaired and checked by reading, not compiled, not run on hardware
//
#include <hip/hip_runtime.h>

// Problem dims (fixed by the reference)
#define B_ 4
#define S_ 1024
#define D_ 1024
#define H_ 4096
#define E_ 8
#define T_ (B_*S_)            // 4096 tokens
#define MTILE 32              // tokens per FFN workgroup
#define MAXTILES (T_/MTILE)   // 128 worst-case tiles per expert

typedef __bf16 bf16x16 __attribute__((ext_vector_type(16)));
typedef __bf16 bf16x2  __attribute__((ext_vector_type(2)));
typedef float  f32x8   __attribute__((ext_vector_type(8)));

union Frag {
  unsigned int u[8];
  uint4        q[2];   // two 16B halves: per-lane contiguous K runs
  bf16x16      v;
};

__device__ __forceinline__ unsigned short f2bf(float f) {
  return __builtin_bit_cast(unsigned short, (__bf16)f);
}
__device__ __forceinline__ unsigned int pack2bf(float lo, float hi) {
  bf16x2 p; p[0] = (__bf16)lo; p[1] = (__bf16)hi;
  return __builtin_bit_cast(unsigned int, p);
}

// ---------------- kernel 0: zero per-expert counters ----------------
__global__ void moe_init(int* counts) {
  if (threadIdx.x < 16) counts[threadIdx.x] = 0;
}

// ---------------- kernel 0b: x f32 -> bf16 (once) ----------------
__global__ __launch_bounds__(256) void xcvt(const float4* __restrict__ x,
                                            uint2* __restrict__ xb, int n4) {
  for (int i = blockIdx.x * blockDim.x + threadIdx.x; i < n4;
       i += gridDim.x * blockDim.x) {
    float4 v = x[i];
    xb[i] = make_uint2(pack2bf(v.x, v.y), pack2bf(v.z, v.w));
  }
}

// ---------------- kernel 0c: weight transpose + f32->bf16 (once) ----------
// src: f32 [E][R][C]; dst: bf16 [E][C][R]  (K becomes the contiguous axis)
__global__ __launch_bounds__(256) void transcvt(const float* __restrict__ src,
                                                unsigned short* __restrict__ dst,
                                                int R, int C) {
  __shared__ float t[32][33];
  const int e = blockIdx.z;
  const float* s = src + (size_t)e * R * C;
  unsigned short* dp = dst + (size_t)e * R * C;
  const int c0 = blockIdx.x * 32, r0 = blockIdx.y * 32;
  const int tx = threadIdx.x;
  for (int i = threadIdx.y; i < 32; i += 8)
    t[i][tx] = s[(size_t)(r0 + i) * C + c0 + tx];
  __syncthreads();
  for (int i = threadIdx.y; i < 32; i += 8)
    dp[(size_t)(c0 + i) * R + r0 + tx] = f2bf(t[tx][i]);
}

// ---------------- kernel 1: gating + top-2 routing ----------------
__global__ __launch_bounds__(256) void moe_gate(
    const float* __restrict__ x, const float* __restrict__ Wg,
    const float* __restrict__ bg, float* __restrict__ gout,
    int* __restrict__ counts, int* __restrict__ lists) {
  const int wave = threadIdx.x >> 5;
  const int lane = threadIdx.x & 31;
  const int t = blockIdx.x * 8 + wave;
  if (t >= T_) return;

  float sc[E_];
#pragma unroll
  for (int e = 0; e < E_; ++e) sc[e] = 0.f;

  for (int d = lane; d < D_; d += 32) {
    float xv = x[(size_t)t * D_ + d];
    const float4* w4 = (const float4*)(Wg + (size_t)d * E_);
    float4 a = w4[0], b = w4[1];
    sc[0] += xv * a.x; sc[1] += xv * a.y; sc[2] += xv * a.z; sc[3] += xv * a.w;
    sc[4] += xv * b.x; sc[5] += xv * b.y; sc[6] += xv * b.z; sc[7] += xv * b.w;
  }
#pragma unroll
  for (int e = 0; e < E_; ++e) {
#pragma unroll
    for (int off = 16; off >= 1; off >>= 1)
      sc[e] += __shfl_xor(sc[e], off, 32);
  }
  if (lane == 0) {
#pragma unroll
    for (int e = 0; e < E_; ++e) sc[e] += bg[e];
#pragma unroll
    for (int e = 0; e < E_; ++e) gout[(size_t)t * E_ + e] = sc[e];
    int i1 = 0;
    for (int e = 1; e < E_; ++e) if (sc[e] > sc[i1]) i1 = e;
    int i2 = (i1 == 0) ? 1 : 0;
    for (int e = 0; e < E_; ++e) if (e != i1 && sc[e] > sc[i2]) i2 = e;
    int p1 = atomicAdd(&counts[i1], 1); lists[i1 * T_ + p1] = t * 2 + 0;
    int p2 = atomicAdd(&counts[i2], 1); lists[i2 * T_ + p2] = t * 2 + 1;
  }
}

// ---------------- kernel 2: fused expert FFN (WMMA bf16) ----------------
// block = 512 threads = 16 wave32; one (expert, 32-token tile) per block.
// All operands pre-converted bf16; fragments load as b128 from LDS/global.
__global__ __launch_bounds__(512, 1) void moe_ffn(
    const unsigned short* __restrict__ xbf,  // [T][D] bf16
    const unsigned short* __restrict__ w1t,  // [E][H][D] bf16 (transposed W1)
    const unsigned short* __restrict__ w2t,  // [E][D][H] bf16 (transposed W2)
    const float* __restrict__ b1, const float* __restrict__ b2,
    const int* __restrict__ counts, const int* __restrict__ lists,
    float* __restrict__ contrib) {
  __shared__ unsigned short xa[MTILE * D_];   // 64 KB  x tile (bf16)
  __shared__ unsigned short hbuf[MTILE * 32]; //  2 KB  h chunk (bf16)
  __shared__ float red[4][4][256];            // 16 KB  split-K partials
  __shared__ int tokrow[MTILE];

  const int e    = blockIdx.x / MAXTILES;
  const int tile = blockIdx.x % MAXTILES;
  const int cnt  = counts[e];
  if (tile * MTILE >= cnt) return;            // uniform early exit

  const int tid   = threadIdx.x;
  const int wave  = tid >> 5;
  const int lane  = tid & 31;
  const int lhalf = (lane < 16) ? 0 : 1;
  const int l15   = lane & 15;

  if (tid < MTILE) {
    int idx = tile * MTILE + tid;
    tokrow[tid] = (idx < cnt) ? lists[e * T_ + idx] : -1;
  }
  __syncthreads();

  // stage gathered bf16 x rows (pure copy, 8B granules)
  for (int i = tid; i < MTILE * D_ / 4; i += 512) {
    int m = i >> 8, o = (i & 255) * 4;
    int entry = tokrow[m];
    uint2 v = (entry >= 0)
        ? *(const uint2*)(xbf + (size_t)(entry >> 1) * D_ + o)
        : make_uint2(0u, 0u);
    *(uint2*)(xa + m * D_ + o) = v;
  }
  __syncthreads();

  const unsigned short* w1te = w1t + (size_t)e * D_ * H_; // [n<H][d<D]
  const unsigned short* w2te = w2t + (size_t)e * H_ * D_; // [n<D][k<H]
  const float* b1e = b1 + (size_t)e * H_;
  const float* b2e = b2 + (size_t)e * D_;

  const int t4 = wave & 3;          // phase-1 output tile (2x2 of 16x16)
  const int mt = t4 >> 1, nt = t4 & 1;
  const int ks = wave >> 2;         // phase-1 K slice: ks*256
  const int n0 = wave * 64;         // phase-2 output N slice

  f32x8 acc[2][4];
#pragma unroll
  for (int a = 0; a < 2; ++a)
#pragma unroll
    for (int b = 0; b < 4; ++b)
#pragma unroll
      for (int r = 0; r < 8; ++r) acc[a][b][r] = 0.f;

  const int kbA   = lhalf * 16;     // 16-bit A-fragment K base per lane half
  const int koffB = lhalf * 16;     // 16-bit B-fragment K base per lane half

  const unsigned short* xrow = xa + (mt * 16 + l15) * D_;

  for (int kc = 0; kc < H_; kc += 32) {
    // ---------- phase 1: hc = relu(x_tile @ W1[:,kc:kc+32] + b1) ----------
    f32x8 c1;
#pragma unroll
    for (int r = 0; r < 8; ++r) c1[r] = 0.f;
    const unsigned short* w1col = w1te + (size_t)(kc + nt * 16 + l15) * D_;
#pragma unroll 2
    for (int step = 0; step < 8; ++step) {
      const int d0 = ks * 256 + step * 32;
      Frag a, b;
      a.q[0] = *(const uint4*)(xrow + d0 + kbA);        // K runs contiguous
      a.q[1] = *(const uint4*)(xrow + d0 + kbA + 8);
      b.q[0] = *(const uint4*)(w1col + d0 + koffB);
      b.q[1] = *(const uint4*)(w1col + d0 + koffB + 8);
      c1 = __builtin_amdgcn_wmma_f32_16x16x32_bf16(false, a.v, false, b.v,
                                                   (short)0, c1, false, false);
    }
#pragma unroll
    for (int r = 0; r < 8; ++r) red[t4][ks][r * 32 + lane] = c1[r];
    __syncthreads();

    if (wave < 4) {                 // reduce 4 K-slices, +b1, relu, cvt bf16
      const int tt = wave, mt2 = tt >> 1, nt2 = tt & 1;
#pragma unroll
      for (int r = 0; r < 8; ++r) {
        float s = red[tt][0][r * 32 + lane] + red[tt][1][r * 32 + lane] +
                  red[tt][2][r * 32 + lane] + red[tt][3][r * 32 + lane];
        int m = mt2 * 16 + lhalf * 8 + r;
        int n = nt2 * 16 + l15;
        s += b1e[kc + n];
        s = s > 0.f ? s : 0.f;
        hbuf[m * 32 + n] = f2bf(s);
      }
    }
    __syncthreads();

    // ---------- phase 2: acc += hc @ W2[kc:kc+32, n0:n0+64] ----------
    Frag a2[2];
#pragma unroll
    for (int m2 = 0; m2 < 2; ++m2) {
      const unsigned short* hrow = hbuf + (m2 * 16 + l15) * 32;
      a2[m2].q[0] = *(const uint4*)(hrow + kbA);
      a2[m2].q[1] = *(const uint4*)(hrow + kbA + 8);
    }
#pragma unroll
    for (int ntile = 0; ntile < 4; ++ntile) {
      const unsigned short* w2col =
          w2te + (size_t)(n0 + ntile * 16 + l15) * H_ + kc + koffB;
      Frag b;
      b.q[0] = *(const uint4*)(w2col);
      b.q[1] = *(const uint4*)(w2col + 8);
      __builtin_prefetch(w2col + 32, 0, 1);   // next H-chunk of W2
#pragma unroll
      for (int m2 = 0; m2 < 2; ++m2)
        acc[m2][ntile] = __builtin_amdgcn_wmma_f32_16x16x32_bf16(
            false, a2[m2].v, false, b.v, (short)0, acc[m2][ntile], false, false);
    }
    __syncthreads();
  }

  // ---------- store per-(token,slot) contributions (+b2), no atomics ------
#pragma unroll
  for (int m2 = 0; m2 < 2; ++m2) {
#pragma unroll
    for (int ntile = 0; ntile < 4; ++ntile) {
      const int col = n0 + ntile * 16 + l15;
      const float bb = b2e[col];
#pragma unroll
      for (int r = 0; r < 8; ++r) {
        int m = m2 * 16 + lhalf * 8 + r;
        int entry = tokrow[m];
        if (entry >= 0) {
          int tok = entry >> 1, slot = entry & 1;
          contrib[((size_t)slot * T_ + tok) * D_ + col] = acc[m2][ntile][r] + bb;
        }
      }
    }
  }
}

// ---------------- kernel 3: combine the two expert slots ----------------
__global__ __launch_bounds__(256) void moe_combine(
    const float4* __restrict__ c0, const float4* __restrict__ c1,
    float4* __restrict__ out, int n4) {
  for (int i = blockIdx.x * blockDim.x + threadIdx.x; i < n4;
       i += gridDim.x * blockDim.x) {
    float4 a = c0[i], b = c1[i];
    out[i] = make_float4(a.x + b.x, a.y + b.y, a.z + b.z, a.w + b.w);
  }
}

extern "C" void kernel_launch(void* const* d_in, const int* in_sizes, int n_in,
                              void* d_out, int out_size, void* d_ws, size_t ws_size,
                              hipStream_t stream) {
  (void)in_sizes; (void)n_in; (void)out_size; (void)ws_size;
  const float* x  = (const float*)d_in[0];
  const float* W1 = (const float*)d_in[1];
  const float* b1 = (const float*)d_in[2];
  const float* W2 = (const float*)d_in[3];
  const float* b2 = (const float*)d_in[4];
  const float* Wg = (const float*)d_in[5];
  const float* bg = (const float*)d_in[6];
  // d_in[7] = top_k (always 2 here)

  float* out  = (float*)d_out;                 // [T, D]
  float* gout = out + (size_t)T_ * D_;         // [T, E] gating scores

  // workspace layout (all chunks >=1KB => alignment preserved)
  char* ws = (char*)d_ws;
  size_t off = 0;
  int* counts = (int*)(ws + off);                    off += 1024;
  int* lists  = (int*)(ws + off);                    off += (size_t)E_ * T_ * 4;
  unsigned short* xbf = (unsigned short*)(ws + off); off += (size_t)T_ * D_ * 2;
  unsigned short* w1t = (unsigned short*)(ws + off); off += (size_t)E_ * D_ * H_ * 2;
  unsigned short* w2t = (unsigned short*)(ws + off); off += (size_t)E_ * H_ * D_ * 2;
  float* contrib = (float*)(ws + off);               // 2*T*D floats

  hipLaunchKernelGGL(moe_init, dim3(1), dim3(32), 0, stream, counts);
  hipLaunchKernelGGL(xcvt, dim3(1024), dim3(256), 0, stream,
                     (const float4*)x, (uint2*)xbf, T_ * D_ / 4);
  hipLaunchKernelGGL(transcvt, dim3(H_ / 32, D_ / 32, E_), dim3(32, 8), 0, stream,
                     W1, w1t, D_, H_);
  hipLaunchKernelGGL(transcvt, dim3(D_ / 32, H_ / 32, E_), dim3(32, 8), 0, stream,
                     W2, w2t, H_, D_);
  hipLaunchKernelGGL(moe_gate, dim3(T_ / 8), dim3(256), 0, stream,
                     x, Wg, bg, gout, counts, lists);
  hipLaunchKernelGGL(moe_ffn, dim3(E_ * MAXTILES), dim3(512), 0, stream,
                     xbf, w1t, w2t, b1, b2, counts, lists, contrib);
  const int n4 = T_ * D_ / 4;
  hipLaunchKernelGGL(moe_combine, dim3(1024), dim3(256), 0, stream,
                     (const float4*)contrib,
                     (const float4*)(contrib + (size_t)T_ * D_),
                     (float4*)out, n4);
}
